// ModDGCNN_37357625540993
// MI455X (gfx1250) — matrix operand order repaired and checked by reading, not compiled, not run on hardware
//
#include <hip/hip_runtime.h>
#include <hip/hip_fp16.h>

typedef _Float16 h16;
typedef __attribute__((ext_vector_type(16))) _Float16 v16h;
typedef __attribute__((ext_vector_type(8)))  float    v8f;

#define KNN 20
#define LRELU(v) ((v) >= 0.0f ? (v) : 0.2f * (v))

// ---------------------------------------------------------------------------
// Weight prep: copy O x I f32 weight -> Np x Kp f16 (row-major, zero padded).
// Kept in native (out x in) layout: WMMA B operand is loaded column-major,
// so lane (= N column) fragments are contiguous rows of this buffer.
// ---------------------------------------------------------------------------
__global__ void k_wprep(const float* __restrict__ W, h16* __restrict__ Wp,
                        int O, int I, int Np, int Kp) {
  long i = (long)blockIdx.x * blockDim.x + threadIdx.x;
  if (i >= (long)Np * Kp) return;
  int k = (int)(i % Kp), n = (int)(i / Kp);
  float v = (n < O && k < I) ? W[(size_t)n * I + k] : 0.0f;
  Wp[i] = (h16)v;
}

__global__ void k_cvt16(const float* __restrict__ s, h16* __restrict__ d, long n) {
  long i = (long)blockIdx.x * blockDim.x + threadIdx.x;
  if (i < n) d[i] = (h16)s[i];
}

// x (B,3,N) -> xb (B,N,3)
__global__ void k_transpose_x(const float* __restrict__ x, float* __restrict__ xb,
                              int B, int C, int N) {
  long i = (long)blockIdx.x * blockDim.x + threadIdx.x;
  if (i >= (long)B * C * N) return;
  int n = (int)(i % N); long t = i / N; int c = (int)(t % C); int b = (int)(t / C);
  xb[((size_t)b * N + n) * C + c] = x[i];
}

// ---------------------------------------------------------------------------
// KNN top-20 by negative squared distance. One wave (32 lanes) per query.
// Candidates staged through LDS in chunks of 64; per-lane sorted top-20
// lists merged with a wave-wide argmax (shfl_xor butterfly, wave32).
// ---------------------------------------------------------------------------
__device__ __forceinline__ bool knn_better(float d, int i, float d2, int i2) {
  return (d > d2) || (d == d2 && i < i2);
}

__global__ __launch_bounds__(256) void k_knn(const float* __restrict__ X, int ldx,
                                             int C, int N, int* __restrict__ idx) {
  __shared__ float sq[8 * 128];    // 8 queries x up to 128 channels
  __shared__ float sch[64 * 128];  // 64 candidates x up to 128 channels
  const int nblk = N / 8;
  int b  = blockIdx.x / nblk;
  int n0 = (blockIdx.x % nblk) * 8;
  int wave = threadIdx.x >> 5, lane = threadIdx.x & 31;
  const float* Xb = X + (size_t)b * N * ldx;
  for (int t = threadIdx.x; t < 8 * C; t += 256)
    sq[t] = Xb[(size_t)(n0 + t / C) * ldx + (t % C)];

  float ld_[KNN]; int li_[KNN];
  for (int k = 0; k < KNN; ++k) { ld_[k] = -3.0e38f; li_[k] = 0x7fffffff; }

  for (int m0 = 0; m0 < N; m0 += 64) {
    __syncthreads();
    for (int t = threadIdx.x; t < 64 * C; t += 256)
      sch[t] = Xb[(size_t)(m0 + t / C) * ldx + (t % C)];
    __syncthreads();
    for (int mm = lane; mm < 64; mm += 32) {
      int m = m0 + mm;
      float d = 0.0f;
      for (int c = 0; c < C; ++c) {
        float df = sq[wave * C + c] - sch[mm * C + c];
        d -= df * df;
      }
      if (knn_better(d, m, ld_[KNN - 1], li_[KNN - 1])) {
        int p = KNN - 1;
        while (p > 0 && knn_better(d, m, ld_[p - 1], li_[p - 1])) {
          ld_[p] = ld_[p - 1]; li_[p] = li_[p - 1]; --p;
        }
        ld_[p] = d; li_[p] = m;
      }
    }
  }
  int p = 0;
  for (int k = 0; k < KNN; ++k) {
    float bd = (p < KNN) ? ld_[p] : -3.0e38f;
    int   bi = (p < KNN) ? li_[p] : 0x7fffffff;
    int   bl = lane;
    for (int o = 16; o > 0; o >>= 1) {
      float od = __shfl_xor(bd, o, 32);
      int   oi = __shfl_xor(bi, o, 32);
      int   ol = __shfl_xor(bl, o, 32);
      if (knn_better(od, oi, bd, bi)) { bd = od; bi = oi; bl = ol; }
    }
    if (lane == bl) ++p;
    if (lane == 0) idx[((size_t)b * N + (n0 + wave)) * KNN + k] = bi;
  }
}

// Edge features for one batch bb: rows (n*20+k), cols [x[m]-x[n], x[n]], padded.
__global__ void k_feat(const float* __restrict__ X, int ldx, const int* __restrict__ idx,
                       h16* __restrict__ A, int N, int C, int Kp, int bb) {
  long i = (long)blockIdx.x * blockDim.x + threadIdx.x;
  if (i >= (long)N * KNN * Kp) return;
  int col = (int)(i % Kp); long r = i / Kp;
  int k = (int)(r % KNN); int n = (int)(r / KNN);
  const float* Xb = X + (size_t)bb * N * ldx;
  float v = 0.0f;
  if (col < 2 * C) {
    if (col < C) {
      int m = idx[((size_t)bb * N + n) * KNN + k];
      v = Xb[(size_t)m * ldx + col] - Xb[(size_t)n * ldx + col];
    } else {
      v = Xb[(size_t)n * ldx + (col - C)];
    }
  }
  A[i] = (h16)v;
}

// ---------------------------------------------------------------------------
// WMMA fragment loaders (layouts per CDNA5 ISA):
//  A 16x32 f16: lane&15 = M row; lanes 0-15 K={0..7,16..23},
//               lanes 16-31 K={8..15,24..31} -> two 16B contiguous loads.
//  B 32x16 f16 (column-major load): lane&15 = N column; K half by lane half.
// ---------------------------------------------------------------------------
struct U32x8 { uint4 lo, hi; };
__device__ __forceinline__ v16h ld_frag(const h16* p0, const h16* p1) {
  U32x8 t;
  t.lo = *(const uint4*)p0;
  t.hi = *(const uint4*)p1;
  return __builtin_bit_cast(v16h, t);
}
__device__ __forceinline__ v8f wmma_f16(v16h a, v16h b, v8f c) {
  return __builtin_amdgcn_wmma_f32_16x16x32_f16(false, a, false, b,
                                                (short)0, c, false, false);
}

// Generic 1 wave -> one 16x16 tile (fallback for tiny M / ragged N).
__global__ __launch_bounds__(32) void k_wmma_gemm(
    const h16* __restrict__ A, const h16* __restrict__ Bw, float* __restrict__ Out,
    int K, int Npad, int ldo, int Mvalid, int Nvalid,
    const float* __restrict__ bias, const float* __restrict__ scale,
    const float* __restrict__ shift, const float* __restrict__ resid, int act) {
  const int lane = threadIdx.x;
  const int ntile = blockIdx.x, mtile = blockIdx.y;
  const bool hi = lane >= 16;
  const int l15 = lane & 15;
  const int arow = mtile * 16 + l15;
  const int kA0 = hi ? 8 : 0, kA1 = hi ? 24 : 16;
  const int bcol = ntile * 16 + l15;
  const int kB = hi ? 8 : 0;
  const h16* Ar = A + (size_t)arow * K;
  const h16* Br = Bw + (size_t)bcol * K + kB;
  v8f acc = {};
  for (int kb = 0; kb < K; kb += 32) {
    v16h av = ld_frag(Ar + kb + kA0, Ar + kb + kA1);
    v16h bv = ld_frag(Br + kb, Br + kb + 16);
    __builtin_prefetch(Ar + kb + 64, 0, 1);
    acc = wmma_f16(av, bv, acc);
  }
  if (bcol < Nvalid) {
    const float inv = rsqrtf(1.0f + 1e-5f);
    float s  = scale ? scale[bcol] * inv : 1.0f;
    float sh = shift ? shift[bcol] : 0.0f;
    float bi = bias ? bias[bcol] : 0.0f;
    const int rbase = mtile * 16 + (hi ? 8 : 0);
#pragma unroll
    for (int r = 0; r < 8; ++r) {
      int row = rbase + r;
      if (row < Mvalid) {
        float v = s * (acc[r] + bi) + sh;
        if (resid) v += resid[(size_t)row * ldo + bcol];
        if (act) v = LRELU(v);
        Out[(size_t)row * ldo + bcol] = v;
      }
    }
  }
}

// Register-blocked GEMM: one wave -> 32x64 output (2 M-tiles x 4 N-tiles),
// 8 WMMAs per 32-wide K step. Requires Mpad%32==0, Npad%64==0.
__global__ __launch_bounds__(32) void k_wmma_gemm_t(
    const h16* __restrict__ A, const h16* __restrict__ Bw, float* __restrict__ Out,
    int K, int ldo, int Mvalid, int Nvalid,
    const float* __restrict__ bias, const float* __restrict__ scale,
    const float* __restrict__ shift, const float* __restrict__ resid, int act) {
  const int lane = threadIdx.x;
  const bool hi = lane >= 16;
  const int l15 = lane & 15;
  const int mbase = blockIdx.y * 32;
  const int nbase = blockIdx.x * 64;
  const int kA0 = hi ? 8 : 0, kA1 = hi ? 24 : 16;
  const int kB = hi ? 8 : 0;
  const h16* Ar0 = A + (size_t)(mbase + l15) * K;
  const h16* Ar1 = A + (size_t)(mbase + 16 + l15) * K;
  const h16* Br  = Bw + (size_t)(nbase + l15) * K + kB;
  const size_t bs = (size_t)16 * K;   // B stride between N-tiles
  v8f z = {};
  v8f acc[2][4];
#pragma unroll
  for (int mi = 0; mi < 2; ++mi)
#pragma unroll
    for (int nj = 0; nj < 4; ++nj) acc[mi][nj] = z;

  for (int kb = 0; kb < K; kb += 32) {
    v16h a0 = ld_frag(Ar0 + kb + kA0, Ar0 + kb + kA1);
    v16h a1 = ld_frag(Ar1 + kb + kA0, Ar1 + kb + kA1);
    v16h b0 = ld_frag(Br + kb,          Br + kb + 16);
    v16h b1 = ld_frag(Br + bs + kb,     Br + bs + kb + 16);
    v16h b2 = ld_frag(Br + 2 * bs + kb, Br + 2 * bs + kb + 16);
    v16h b3 = ld_frag(Br + 3 * bs + kb, Br + 3 * bs + kb + 16);
    __builtin_prefetch(Ar0 + kb + 128, 0, 1);
    __builtin_prefetch(Ar1 + kb + 128, 0, 1);
    acc[0][0] = wmma_f16(a0, b0, acc[0][0]);
    acc[0][1] = wmma_f16(a0, b1, acc[0][1]);
    acc[0][2] = wmma_f16(a0, b2, acc[0][2]);
    acc[0][3] = wmma_f16(a0, b3, acc[0][3]);
    acc[1][0] = wmma_f16(a1, b0, acc[1][0]);
    acc[1][1] = wmma_f16(a1, b1, acc[1][1]);
    acc[1][2] = wmma_f16(a1, b2, acc[1][2]);
    acc[1][3] = wmma_f16(a1, b3, acc[1][3]);
  }

  const float inv = rsqrtf(1.0f + 1e-5f);
#pragma unroll
  for (int nj = 0; nj < 4; ++nj) {
    int col = nbase + nj * 16 + l15;
    if (col >= Nvalid) continue;
    float s  = scale ? scale[col] * inv : 1.0f;
    float sh = shift ? shift[col] : 0.0f;
    float bi = bias ? bias[col] : 0.0f;
#pragma unroll
    for (int mi = 0; mi < 2; ++mi) {
      const int rbase = mbase + mi * 16 + (hi ? 8 : 0);
#pragma unroll
      for (int r = 0; r < 8; ++r) {
        int row = rbase + r;
        if (row < Mvalid) {
          float v = s * (acc[mi][nj][r] + bi) + sh;
          if (resid) v += resid[(size_t)row * ldo + col];
          if (act) v = LRELU(v);
          Out[(size_t)row * ldo + col] = v;
        }
      }
    }
  }
}

// BN + LReLU + max over the 20 neighbors (bn1_g > 0 in reference, order-safe).
__global__ void k_redmax(const float* __restrict__ T, float* __restrict__ H,
                         int N, int Cout, const float* __restrict__ g,
                         const float* __restrict__ bshift, int bb) {
  long i = (long)blockIdx.x * blockDim.x + threadIdx.x;
  if (i >= (long)N * Cout) return;
  int c = (int)(i % Cout), n = (int)(i / Cout);
  float s = g[c] * rsqrtf(1.0f + 1e-5f), sh = bshift[c];
  float best = -3.0e38f;
  const float* row = T + (size_t)n * KNN * Cout + c;
  for (int k = 0; k < KNN; ++k) {
    float v = s * row[(size_t)k * Cout] + sh;
    v = LRELU(v);
    best = fmaxf(best, v);
  }
  H[((size_t)bb * N + n) * Cout + c] = best;
}

// Tiny attention over the batch dim (8x8 per (n,head)); writes o as f16.
__global__ void k_attn(const float* __restrict__ qkv, h16* __restrict__ o16,
                       int Cout, int N) {
  long i = (long)blockIdx.x * blockDim.x + threadIdx.x;
  if (i >= (long)N * 8) return;
  int h = (int)(i % 8), n = (int)(i / 8);
  int dh = Cout / 8;
  float sc[8][8];
  float qs = rsqrtf((float)dh);
  for (int qb = 0; qb < 8; ++qb) {
    const float* qp = qkv + ((size_t)qb * N + n) * 3 * Cout + h * dh;
    for (int kb = 0; kb < 8; ++kb) {
      const float* kp = qkv + ((size_t)kb * N + n) * 3 * Cout + Cout + h * dh;
      float s = 0.0f;
      for (int d = 0; d < dh; ++d) s += qp[d] * kp[d];
      sc[qb][kb] = s * qs;
    }
  }
  for (int qb = 0; qb < 8; ++qb) {
    float mx = sc[qb][0];
    for (int kb = 1; kb < 8; ++kb) mx = fmaxf(mx, sc[qb][kb]);
    float sm = 0.0f;
    for (int kb = 0; kb < 8; ++kb) { sc[qb][kb] = __expf(sc[qb][kb] - mx); sm += sc[qb][kb]; }
    float rr = 1.0f / sm;
    for (int kb = 0; kb < 8; ++kb) sc[qb][kb] *= rr;
  }
  for (int qb = 0; qb < 8; ++qb)
    for (int d = 0; d < dh; ++d) {
      float a = 0.0f;
      for (int kb = 0; kb < 8; ++kb)
        a += sc[qb][kb] * qkv[((size_t)kb * N + n) * 3 * Cout + 2 * Cout + h * dh + d];
      o16[((size_t)qb * N + n) * Cout + h * dh + d] = (h16)a;
    }
}

// concat [h_mha(Cout), x_in(Cin)] -> f16 padded to Kp.
__global__ void k_concat(const float* __restrict__ Hm, const float* __restrict__ Xin, int ldx,
                         h16* __restrict__ A, int Cout, int Cin, int Kp, long rows) {
  long i = (long)blockIdx.x * blockDim.x + threadIdx.x;
  if (i >= rows * Kp) return;
  int col = (int)(i % Kp); long r = i / Kp;
  float v = 0.0f;
  if (col < Cout) v = Hm[(size_t)r * Cout + col];
  else if (col < Cout + Cin) v = Xin[(size_t)r * ldx + (col - Cout)];
  A[i] = (h16)v;
}

// global max + mean pool over N -> g16 (rows 0..7 valid of padded 16)
__global__ void k_pool(const float* __restrict__ xc, h16* __restrict__ g16, int N, int C) {
  long i = (long)blockIdx.x * blockDim.x + threadIdx.x;
  if (i >= (long)8 * C) return;
  int b = (int)(i / C), c = (int)(i % C);
  float mx = -3.0e38f, sm = 0.0f;
  for (int n = 0; n < N; ++n) {
    float v = xc[((size_t)b * N + n) * C + c];
    mx = fmaxf(mx, v); sm += v;
  }
  g16[(size_t)b * 2 * C + c]     = (h16)mx;
  g16[(size_t)b * 2 * C + C + c] = (h16)(sm / (float)N);
}

// ---------------------------------------------------------------------------
extern "C" void kernel_launch(void* const* d_in, const int* in_sizes, int n_in,
                              void* d_out, int out_size, void* d_ws, size_t ws_size,
                              hipStream_t stream) {
  (void)in_sizes; (void)n_in; (void)out_size; (void)ws_size;
  const int B = 8, N = 2048, BN = B * N;
  const int cin_[4]  = {3, 64, 64, 128};
  const int cout_[4] = {64, 64, 128, 256};
  const int convKp[4] = {32, 128, 128, 256};   // pad32(2*cin)
  const int skipKp[4] = {96, 128, 192, 384};   // pad32(cin+cout)
  const int foff[4]   = {0, 64, 128, 256};     // feature concat offsets

  int ii = 0;
  const float* x = (const float*)d_in[ii++];
  const float *conv_w[4], *bn1_g[4], *bn1_b[4], *w_in[4], *b_in[4],
              *w_out[4], *b_out[4], *skip_w[4], *bn2_g[4], *bn2_b[4];
  for (int b = 0; b < 4; ++b) {
    conv_w[b] = (const float*)d_in[ii++]; bn1_g[b] = (const float*)d_in[ii++];
    bn1_b[b] = (const float*)d_in[ii++];  w_in[b] = (const float*)d_in[ii++];
    b_in[b] = (const float*)d_in[ii++];   w_out[b] = (const float*)d_in[ii++];
    b_out[b] = (const float*)d_in[ii++];  skip_w[b] = (const float*)d_in[ii++];
    bn2_g[b] = (const float*)d_in[ii++];  bn2_b[b] = (const float*)d_in[ii++];
  }
  const float* conv5_w = (const float*)d_in[ii++];
  const float* bn5_g = (const float*)d_in[ii++];
  const float* bn5_b = (const float*)d_in[ii++];
  const float* l1_w  = (const float*)d_in[ii++];
  const float* bn6_g = (const float*)d_in[ii++];
  const float* bn6_b = (const float*)d_in[ii++];
  const float* l2_w  = (const float*)d_in[ii++];
  const float* l2_b  = (const float*)d_in[ii++];
  const float* bn7_g = (const float*)d_in[ii++];
  const float* bn7_b = (const float*)d_in[ii++];
  const float* l3_w  = (const float*)d_in[ii++];
  const float* l3_b  = (const float*)d_in[ii++];

  // ---- workspace layout ----
  char* base = (char*)d_ws; size_t off = 0;
  auto alloc = [&](size_t bytes) -> void* {
    off = (off + 255) & ~(size_t)255;
    void* p = base + off; off += bytes; return p;
  };
  float* xb = (float*)alloc((size_t)BN * 3 * 4);
  h16 *convw16[4], *win16[4], *wout16[4], *skipw16[4];
  for (int b = 0; b < 4; ++b) {
    convw16[b] = (h16*)alloc((size_t)cout_[b] * convKp[b] * 2);
    win16[b]   = (h16*)alloc((size_t)3 * cout_[b] * cout_[b] * 2);
    wout16[b]  = (h16*)alloc((size_t)cout_[b] * cout_[b] * 2);
    skipw16[b] = (h16*)alloc((size_t)cout_[b] * skipKp[b] * 2);
  }
  h16* conv5_16 = (h16*)alloc((size_t)1024 * 512 * 2);
  h16* l1_16 = (h16*)alloc((size_t)512 * 2048 * 2);
  h16* l2_16 = (h16*)alloc((size_t)256 * 512 * 2);
  h16* l3_16 = (h16*)alloc((size_t)48 * 256 * 2);
  int* idx = (int*)alloc((size_t)BN * KNN * 4);
  h16* feat16 = (h16*)alloc((size_t)N * KNN * 256 * 2);  // per-batch edge features
  float* gtmp = (float*)alloc((size_t)N * KNN * 256 * 4); // per-batch conv output
  float* hA = (float*)alloc((size_t)BN * 256 * 4);        // pre-MHA block features
  h16* hA16 = (h16*)alloc((size_t)BN * 256 * 2);
  float* qkvb = (float*)alloc((size_t)BN * 768 * 4);
  h16* o16 = (h16*)alloc((size_t)BN * 256 * 2);
  float* mhab = (float*)alloc((size_t)BN * 256 * 4);
  h16* catA = (h16*)alloc((size_t)BN * 384 * 2);
  float* feats = (float*)alloc((size_t)BN * 512 * 4);     // concat of block outputs
  h16* fA16 = (h16*)alloc((size_t)BN * 512 * 2);
  float* xc = (float*)alloc((size_t)BN * 1024 * 4);
  h16* g16 = (h16*)alloc((size_t)16 * 2048 * 2);
  float* l1o = (float*)alloc((size_t)16 * 512 * 4);
  h16* l1o16 = (h16*)alloc((size_t)16 * 512 * 2);
  float* l2o = (float*)alloc((size_t)16 * 256 * 4);
  h16* l2o16 = (h16*)alloc((size_t)16 * 256 * 2);

  auto cdiv = [](long a, long b) { return (int)((a + b - 1) / b); };
  auto wprep = [&](const float* W, h16* Wp, int O, int I, int Np, int Kp) {
    k_wprep<<<cdiv((long)Np * Kp, 256), 256, 0, stream>>>(W, Wp, O, I, Np, Kp);
  };
  auto gemm = [&](const h16* A, const h16* Bw, float* Out, int Mpad, int Mvalid,
                  int Npad, int Nvalid, int K, int ldo,
                  const float* bias, const float* scale, const float* shift,
                  const float* resid, int act) {
    if (Mpad % 32 == 0 && Npad % 64 == 0) {
      dim3 g(Npad / 64, Mpad / 32);
      k_wmma_gemm_t<<<g, dim3(32), 0, stream>>>(A, Bw, Out, K, ldo, Mvalid,
                                                Nvalid, bias, scale, shift, resid, act);
    } else {
      dim3 g(Npad / 16, Mpad / 16);
      k_wmma_gemm<<<g, dim3(32), 0, stream>>>(A, Bw, Out, K, Npad, ldo, Mvalid,
                                              Nvalid, bias, scale, shift, resid, act);
    }
  };

  // ---- weights -> padded f16 ----
  k_transpose_x<<<cdiv((long)B * 3 * N, 256), 256, 0, stream>>>(x, xb, B, 3, N);
  for (int b = 0; b < 4; ++b) {
    wprep(conv_w[b], convw16[b], cout_[b], 2 * cin_[b], cout_[b], convKp[b]);
    wprep(w_in[b], win16[b], 3 * cout_[b], cout_[b], 3 * cout_[b], cout_[b]);
    wprep(w_out[b], wout16[b], cout_[b], cout_[b], cout_[b], cout_[b]);
    wprep(skip_w[b], skipw16[b], cout_[b], cin_[b] + cout_[b], cout_[b], skipKp[b]);
  }
  wprep(conv5_w, conv5_16, 1024, 512, 1024, 512);
  wprep(l1_w, l1_16, 512, 2048, 512, 2048);
  wprep(l2_w, l2_16, 256, 512, 256, 512);
  wprep(l3_w, l3_16, 40, 256, 48, 256);

  // ---- EdgeConv blocks ----
  for (int b = 0; b < 4; ++b) {
    const float* X = (b == 0) ? xb : (feats + foff[b - 1]);
    const int ldx = (b == 0) ? 3 : 512;
    const int ci = cin_[b], co = cout_[b], Kp = convKp[b];

    k_knn<<<B * (N / 8), 256, 0, stream>>>(X, ldx, ci, N, idx);
    for (int bb = 0; bb < B; ++bb) {
      k_feat<<<cdiv((long)N * KNN * Kp, 256), 256, 0, stream>>>(X, ldx, idx, feat16, N, ci, Kp, bb);
      gemm(feat16, convw16[b], gtmp, N * KNN, N * KNN, co, co, Kp, co,
           nullptr, nullptr, nullptr, nullptr, 0);
      k_redmax<<<cdiv((long)N * co, 256), 256, 0, stream>>>(gtmp, hA, N, co, bn1_g[b], bn1_b[b], bb);
    }
    k_cvt16<<<cdiv((long)BN * co, 256), 256, 0, stream>>>(hA, hA16, (long)BN * co);
    gemm(hA16, win16[b], qkvb, BN, BN, 3 * co, 3 * co, co, 3 * co,
         b_in[b], nullptr, nullptr, nullptr, 0);
    k_attn<<<cdiv((long)N * 8, 256), 256, 0, stream>>>(qkvb, o16, co, N);
    gemm(o16, wout16[b], mhab, BN, BN, co, co, co, co,
         b_out[b], nullptr, nullptr, hA, 0);  // + residual h
    k_concat<<<cdiv((long)BN * skipKp[b], 256), 256, 0, stream>>>(mhab, X, ldx, catA,
                                                                  co, ci, skipKp[b], (long)BN);
    gemm(catA, skipw16[b], feats + foff[b], BN, BN, co, co, skipKp[b], 512,
         nullptr, bn2_g[b], bn2_b[b], nullptr, 1);
  }

  // ---- head ----
  k_cvt16<<<cdiv((long)BN * 512, 256), 256, 0, stream>>>(feats, fA16, (long)BN * 512);
  gemm(fA16, conv5_16, xc, BN, BN, 1024, 1024, 512, 1024,
       nullptr, bn5_g, bn5_b, nullptr, 1);
  k_pool<<<cdiv((long)8 * 1024, 256), 256, 0, stream>>>(xc, g16, N, 1024);
  gemm(g16, l1_16, l1o, 16, 8, 512, 512, 2048, 512, nullptr, bn6_g, bn6_b, nullptr, 1);
  k_cvt16<<<cdiv((long)16 * 512, 256), 256, 0, stream>>>(l1o, l1o16, 16 * 512);
  gemm(l1o16, l2_16, l2o, 16, 8, 256, 256, 512, 256, l2_b, bn7_g, bn7_b, nullptr, 1);
  k_cvt16<<<cdiv((long)16 * 256, 256), 256, 0, stream>>>(l2o, l2o16, 16 * 256);
  gemm(l2o16, l3_16, (float*)d_out, 16, 8, 48, 40, 256, 40,
       l3_b, nullptr, nullptr, nullptr, 0);
}
